// RNN_Model_81501299409182
// MI455X (gfx1250) — compile-verified, matrix operand order checked by reading, not run on hardware
//
#include <hip/hip_runtime.h>
#include <hip/hip_bf16.h>
#include <math.h>

typedef __bf16 bf16_t;
typedef __attribute__((ext_vector_type(16))) __bf16 v16bf;
typedef __attribute__((ext_vector_type(8)))  __bf16 v8bf;
typedef __attribute__((ext_vector_type(8)))  float  v8f;
typedef __attribute__((ext_vector_type(4)))  float  v4f;
typedef int v4i_gcc __attribute__((vector_size(16)));   // matches builtin param type

#define WMMA_BF16(a, b, c) \
  __builtin_amdgcn_wmma_f32_16x16x32_bf16(false, (a), false, (b), (short)0, (c), false, false)

#if defined(__AMDGCN__) && __has_builtin(__builtin_amdgcn_global_load_async_to_lds_b128)
#define HAVE_ASYNC_LDS 1
#else
#define HAVE_ASYNC_LDS 0
#endif

static __device__ __forceinline__ v8f zero_v8f() {
  v8f z;
#pragma unroll
  for (int i = 0; i < 8; ++i) z[i] = 0.0f;
  return z;
}

// A fragment (16x32 bf16) for v_wmma_f32_16x16x32_bf16 from row-major fp32.
// Lane L holds row m0 + (L%16); K runs [kb, kb+8) and [kb+16, kb+24), kb = k0 + (L/16)*8.
static __device__ __forceinline__ v16bf load_a_f32(const float* __restrict__ A, int lda,
                                                   int m0, int k0) {
  const int lane = threadIdx.x & 31;
  const float* p = A + (size_t)(m0 + (lane & 15)) * lda + (k0 + ((lane >> 4) << 3));
  v4f f0 = *(const v4f*)(p);
  v4f f1 = *(const v4f*)(p + 4);
  v4f f2 = *(const v4f*)(p + 16);
  v4f f3 = *(const v4f*)(p + 20);
  v16bf r;
#pragma unroll
  for (int i = 0; i < 4; ++i) {
    r[i]      = (__bf16)f0[i];
    r[i + 4]  = (__bf16)f1[i];
    r[i + 8]  = (__bf16)f2[i];
    r[i + 12] = (__bf16)f3[i];
  }
  return r;
}

// Generic bf16 fragment loader from a row-major [R][K] bf16 matrix (global or LDS).
// For A: rows are M; for B (pre-transposed weights stored [N][K]): rows are N.
// Lane L holds row r0 + (L%16); two contiguous 8-elem K runs at kb and kb+16.
static __device__ __forceinline__ v16bf load_frag_bf16(const bf16_t* base, int ld,
                                                       int r0, int k0) {
  const int lane = threadIdx.x & 31;
  const bf16_t* p = base + (size_t)(r0 + (lane & 15)) * ld + (k0 + ((lane >> 4) << 3));
  v8bf lo = *(const v8bf*)(p);
  v8bf hi = *(const v8bf*)(p + 16);
  v16bf r;
#pragma unroll
  for (int i = 0; i < 8; ++i) { r[i] = lo[i]; r[i + 8] = hi[i]; }
  return r;
}

// Stage a [rows][rowElems] bf16 strip (row-major, pitch srcLd) into LDS with a
// padded pitch (conflict-free ds_load_b128 across 16 lanes). Uses CDNA5 async
// global->LDS DMA when available; tracked with ASYNCcnt.
static __device__ __forceinline__ void stage_strip(const bf16_t* gsrc, int srcLd,
                                                   bf16_t* lds, int dstLd,
                                                   int rows, int rowElems) {
  const int chunksPerRow = rowElems >> 3;          // 16-byte chunks
  const int total = rows * chunksPerRow;
  for (int i = threadIdx.x; i < total; i += blockDim.x) {
    const int r = i / chunksPerRow;
    const int c = i - r * chunksPerRow;
    const bf16_t* g = gsrc + (size_t)r * srcLd + (c << 3);
    bf16_t* l = lds + r * dstLd + (c << 3);
#if HAVE_ASYNC_LDS
    __builtin_amdgcn_global_load_async_to_lds_b128(
        (__attribute__((address_space(1))) v4i_gcc*)const_cast<bf16_t*>(g),
        (__attribute__((address_space(3))) v4i_gcc*)l, 0, 0);
#else
    *(v8bf*)l = *(const v8bf*)g;
#endif
  }
#if HAVE_ASYNC_LDS
#if __has_builtin(__builtin_amdgcn_s_wait_asynccnt)
  __builtin_amdgcn_s_wait_asynccnt(0);
#else
  asm volatile("s_wait_asynccnt 0" ::: "memory");
#endif
#endif
  __syncthreads();
}

// ---------------------------------------------------------------------------
// Prep: W_in (256x512) -> WinT bf16 [512][256], W_h (512x512) -> WhT bf16
// [512][512] (both transposed so B fragments are contiguous), zero H0 (bf16).
// ---------------------------------------------------------------------------
__global__ __launch_bounds__(256) void prep_kernel(const float* __restrict__ Win,
                                                   const float* __restrict__ Wh,
                                                   bf16_t* __restrict__ WinT,
                                                   bf16_t* __restrict__ WhT,
                                                   bf16_t* __restrict__ H0) {
  const int idx = blockIdx.x * 256 + threadIdx.x;   // 0 .. 262143
  if (idx < 512 * 256) {                            // WinT[u][d] = Win[d][u]
    const int u = idx >> 8, d = idx & 255;
    WinT[idx] = (__bf16)Win[d * 512 + u];
  }
  {                                                 // WhT[n][k] = Wh[k][n]
    const int n = idx >> 9, k = idx & 511;
    WhT[idx] = (__bf16)Wh[k * 512 + n];
  }
  if (idx < 256 * 512) H0[idx] = (__bf16)0.0f;
}

// ---------------------------------------------------------------------------
// GEMM1: xproj[b*T+t][u] = bf16( data[row][:] . W_in[:][u] + b_rnn[u] )
// M = 131072, N = 512, K = 256.  Block = 8 waves, wave tile 32x32 (2x2 WMMA),
// block tile 128x64.  Grid (1024, 8).  B strip staged in LDS (async DMA).
// ---------------------------------------------------------------------------
#define G1_PITCH 264   // 256 + 8 bf16 pad -> 528B rows, +4 dword bank skew
__global__ __launch_bounds__(256) void xproj_gemm(const float* __restrict__ data,
                                                  const bf16_t* __restrict__ WinT,
                                                  const float* __restrict__ b_rnn,
                                                  bf16_t* __restrict__ xproj) {
  const int D = 256, U = 512;
  extern __shared__ char smem_raw[];
  bf16_t* bs = (bf16_t*)smem_raw;                   // [64][G1_PITCH]

  stage_strip(WinT + (size_t)(blockIdx.y * 64) * D, D, bs, G1_PITCH, 64, D);

  const int w  = threadIdx.x >> 5;
  const int m0 = blockIdx.x * 128 + (w >> 1) * 32;
  const int nl = (w & 1) * 32;                      // n offset within LDS strip
  const int n0 = blockIdx.y * 64 + nl;

  v8f c00 = zero_v8f(), c01 = zero_v8f(), c10 = zero_v8f(), c11 = zero_v8f();

  for (int k0 = 0; k0 < D; k0 += 32) {
    v16bf a0 = load_a_f32(data, D, m0, k0);
    v16bf a1 = load_a_f32(data, D, m0 + 16, k0);
    v16bf b0 = load_frag_bf16(bs, G1_PITCH, nl, k0);
    v16bf b1 = load_frag_bf16(bs, G1_PITCH, nl + 16, k0);
    c00 = WMMA_BF16(a0, b0, c00);
    c01 = WMMA_BF16(a0, b1, c01);
    c10 = WMMA_BF16(a1, b0, c10);
    c11 = WMMA_BF16(a1, b1, c11);
  }

  const int lane = threadIdx.x & 31;
  const int nA   = n0 + (lane & 15);
  const int mOff = (lane >> 4) << 3;
  const float biasA = b_rnn[nA];
  const float biasB = b_rnn[nA + 16];
  bf16_t* p0 = xproj + (size_t)(m0 + mOff) * U + nA;
  bf16_t* p1 = p0 + (size_t)16 * U;
#pragma unroll
  for (int r = 0; r < 8; ++r) {
    p0[r * U]      = (__bf16)(c00[r] + biasA);
    p0[r * U + 16] = (__bf16)(c01[r] + biasB);
    p1[r * U]      = (__bf16)(c10[r] + biasA);
    p1[r * U + 16] = (__bf16)(c11[r] + biasB);
  }
}

// ---------------------------------------------------------------------------
// Recurrent step: Hnext = bf16( tanh( xproj[:, t, :] + Hprev @ W_h ) )
// M = 256, N = 512, K = 512.  State H lives in bf16 -> A fragments load with
// zero conversion VALU.  Block = 8 waves (2 M x 4 N), tile 64x128, grid (4,4).
// W_h strip (128 x 512 bf16) staged in LDS (async DMA).
// ---------------------------------------------------------------------------
#define ST_PITCH 520   // 512 + 8 bf16 pad -> 1040B rows, +4 dword bank skew
__global__ __launch_bounds__(256) void rnn_step(const bf16_t* __restrict__ Hprev,
                                                const bf16_t* __restrict__ WhT,
                                                const bf16_t* __restrict__ xproj,
                                                bf16_t* __restrict__ Hnext,
                                                int t) {
  const int U = 512, T = 512;
  extern __shared__ char smem_raw[];
  bf16_t* bs = (bf16_t*)smem_raw;                   // [128][ST_PITCH]

  stage_strip(WhT + (size_t)(blockIdx.y * 128) * U, U, bs, ST_PITCH, 128, U);

  const int w  = threadIdx.x >> 5;
  const int m0 = blockIdx.x * 64 + (w >> 2) * 32;
  const int nl = (w & 3) * 32;
  const int n0 = blockIdx.y * 128 + nl;

  v8f c00 = zero_v8f(), c01 = zero_v8f(), c10 = zero_v8f(), c11 = zero_v8f();

  for (int k0 = 0; k0 < U; k0 += 32) {
    v16bf a0 = load_frag_bf16(Hprev, U, m0, k0);
    v16bf a1 = load_frag_bf16(Hprev, U, m0 + 16, k0);
    v16bf b0 = load_frag_bf16(bs, ST_PITCH, nl, k0);
    v16bf b1 = load_frag_bf16(bs, ST_PITCH, nl + 16, k0);
    c00 = WMMA_BF16(a0, b0, c00);
    c01 = WMMA_BF16(a0, b1, c01);
    c10 = WMMA_BF16(a1, b0, c10);
    c11 = WMMA_BF16(a1, b1, c11);
  }

  const int lane = threadIdx.x & 31;
  const int nA   = n0 + (lane & 15);
  const int mOff = (lane >> 4) << 3;
  const bf16_t* x0 = xproj + ((size_t)(m0 + mOff) * T + t) * U + nA;
  const bf16_t* x1 = x0 + (size_t)16 * T * U;
  bf16_t* h0 = Hnext + (size_t)(m0 + mOff) * U + nA;
  bf16_t* h1 = h0 + (size_t)16 * U;
#pragma unroll
  for (int r = 0; r < 8; ++r) {
    h0[r * U]      = (__bf16)tanhf(c00[r] + (float)x0[r * (T * U)]);
    h0[r * U + 16] = (__bf16)tanhf(c01[r] + (float)x0[r * (T * U) + 16]);
    h1[r * U]      = (__bf16)tanhf(c10[r] + (float)x1[r * (T * U)]);
    h1[r * U + 16] = (__bf16)tanhf(c11[r] + (float)x1[r * (T * U) + 16]);
  }
}

// ---------------------------------------------------------------------------
// Head: h = relu(H @ W_d + b_d); logits = h @ W_o + b_o; softmax.
// One block per batch row, 128 threads. Tiny (17 MFLOP) -> scalar VALU.
// ---------------------------------------------------------------------------
__global__ __launch_bounds__(128) void head_kernel(const bf16_t* __restrict__ H,
                                                   const float* __restrict__ Wd,
                                                   const float* __restrict__ bd,
                                                   const float* __restrict__ Wo,
                                                   const float* __restrict__ bo,
                                                   float* __restrict__ out) {
  __shared__ float h[128];
  __shared__ float logits[10];
  const int row = blockIdx.x;
  const int j = threadIdx.x;

  float acc = bd[j];
  const bf16_t* Hrow = H + (size_t)row * 512;
  for (int k = 0; k < 512; ++k) acc += (float)Hrow[k] * Wd[k * 128 + j];
  h[j] = fmaxf(acc, 0.0f);
  __syncthreads();

  if (j < 10) {
    float l = bo[j];
    for (int k = 0; k < 128; ++k) l += h[k] * Wo[k * 10 + j];
    logits[j] = l;
  }
  __syncthreads();

  if (j == 0) {
    float mx = logits[0];
    for (int i = 1; i < 10; ++i) mx = fmaxf(mx, logits[i]);
    float e[10], s = 0.0f;
    for (int i = 0; i < 10; ++i) { e[i] = expf(logits[i] - mx); s += e[i]; }
    const float inv = 1.0f / s;
    for (int i = 0; i < 10; ++i) out[(size_t)row * 10 + i] = e[i] * inv;
  }
}

// ---------------------------------------------------------------------------
// Workspace layout (bytes):
//   [0)          WinT bf16   512*256*2 = 262144
//   [262144)     WhT  bf16   512*512*2 = 524288
//   [786432)     xproj bf16  131072*512*2 = 134217728
//   [135004160)  H0 bf16     256*512*2 = 262144
//   [135266304)  H1 bf16     256*512*2 = 262144
//   total ~135.5 MB
// ---------------------------------------------------------------------------
extern "C" void kernel_launch(void* const* d_in, const int* in_sizes, int n_in,
                              void* d_out, int out_size, void* d_ws, size_t ws_size,
                              hipStream_t stream) {
  (void)in_sizes; (void)n_in; (void)out_size; (void)ws_size;

  const float* data  = (const float*)d_in[0];  // [256,512,256]
  const float* W_in  = (const float*)d_in[1];  // [256,512]
  const float* W_h   = (const float*)d_in[2];  // [512,512]
  const float* b_rnn = (const float*)d_in[3];  // [512]
  const float* W_d   = (const float*)d_in[4];  // [512,128]
  const float* b_d   = (const float*)d_in[5];  // [128]
  const float* W_o   = (const float*)d_in[6];  // [128,10]
  const float* b_o   = (const float*)d_in[7];  // [10]
  float* out = (float*)d_out;                  // [256,10] f32

  char* ws = (char*)d_ws;
  bf16_t* WinT  = (bf16_t*)(ws);
  bf16_t* WhT   = (bf16_t*)(ws + 262144);
  bf16_t* xproj = (bf16_t*)(ws + 786432);
  bf16_t* H0    = (bf16_t*)(ws + 135004160);
  bf16_t* H1    = (bf16_t*)(ws + 135266304);

  // 1) weight transpose/convert + zero initial state
  prep_kernel<<<1024, 256, 0, stream>>>(W_in, W_h, WinT, WhT, H0);

  // 2) x_proj = data @ W_in + b_rnn  (bf16 WMMA, B strip in LDS)
  xproj_gemm<<<dim3(1024, 8), 256, 64 * G1_PITCH * 2, stream>>>(data, WinT, b_rnn, xproj);

  // 3) serial scan: 512 dependent WMMA GEMM steps, bf16 state ping-pong
  bf16_t* Ha = H0;
  bf16_t* Hb = H1;
  for (int t = 0; t < 512; ++t) {
    rnn_step<<<dim3(4, 4), 256, 128 * ST_PITCH * 2, stream>>>(Ha, WhT, xproj, Hb, t);
    bf16_t* tmp = Ha; Ha = Hb; Hb = tmp;
  }
  // T even -> final state back in H0 (== Ha)

  // 4) dense head + softmax
  head_kernel<<<256, 128, 0, stream>>>(Ha, W_d, b_d, W_o, b_o, out);
}
